// TransformerBatchNormDecoderLayer_53867479827113
// MI455X (gfx1250) — compile-verified, matrix operand order checked by reading, not run on hardware
//
#include <hip/hip_runtime.h>
#include <hip/hip_bf16.h>
#include <math.h>
#include <stdint.h>

// ---------------------------------------------------------------------------
// Problem constants (match reference)
// ---------------------------------------------------------------------------
#define BATCH   4
#define SEQ     1024
#define DM      1024
#define NH      16
#define DK      64
#define DFF_N   4096
#define ROWS    (BATCH*SEQ)        // 4096 tokens
#define SAMPLEK 35                 // 5*ceil(ln(1024))
#define NTOP    35

// WMMA types
typedef __attribute__((ext_vector_type(16))) __bf16 v16bf;
typedef __attribute__((ext_vector_type(8)))  __bf16 bf16x8;
typedef __attribute__((ext_vector_type(4)))  __bf16 bf16x4;
typedef __attribute__((ext_vector_type(8)))  float  v8f;

// ---------------------------------------------------------------------------
// Async Global->LDS copy (CDNA5 ASYNCcnt path) + wait
// ---------------------------------------------------------------------------
__device__ __forceinline__ void async_b128(unsigned lds_byte_off, const void* gaddr) {
  // GLOBAL_LOAD_ASYNC_TO_LDS_B128: vdst = LDS byte address, vaddr = 64b global
  asm volatile("global_load_async_to_lds_b128 %0, %1, off"
               :: "v"(lds_byte_off), "v"(gaddr) : "memory");
}
__device__ __forceinline__ void wait_async0() {
#if __has_builtin(__builtin_amdgcn_s_wait_asynccnt)
  __builtin_amdgcn_s_wait_asynccnt(0);
#else
  asm volatile("s_wait_asynccnt 0" ::: "memory");
#endif
}
__device__ __forceinline__ unsigned lds_off(const void* p) {
  // LDS aperture: low 32 bits of the pointer give the LDS byte offset
  return (unsigned)(uintptr_t)p;
}

// ---------------------------------------------------------------------------
// WMMA GEMM: C[M,N] = act(A[M,K] @ BT[N,K]^T + bias)
// A, BT pre-converted to bf16 with contiguous K; fp32 accumulate.
// Block tile 128x128, K-chunk 32. 128 threads = 4 wave32, each wave a 64x64
// tile (4x4 WMMA sub-tiles -> 16 WMMA per k-chunk per wave, 2 WMMA / 32B LDS).
// Double-buffered async Global->LDS staging.
// ---------------------------------------------------------------------------
#define TILE_M 128
#define TILE_N 128
#define TILE_K 32
#define LDSS   40   // bf16 stride: 80B rows -> 16B aligned chunks, spread banks

__device__ __forceinline__ v16bf load_frag(const __bf16* p) {
  // fragment = two contiguous 8-elem (16B) runs: [0..7] and [+16..+23]
  bf16x8 lo = *(const bf16x8*)(p);
  bf16x8 hi = *(const bf16x8*)(p + 16);
  v16bf f;
#pragma unroll
  for (int i = 0; i < 8; ++i) { f[i] = lo[i]; f[i + 8] = hi[i]; }
  return f;
}

__global__ __launch_bounds__(128) void gemm_bf16_wmma(
    const __bf16* __restrict__ A,   // [M][K] bf16
    const __bf16* __restrict__ BT,  // [N][K] bf16 (weights pre-transposed)
    const float* __restrict__ bias, float* __restrict__ C,
    int M, int N, int K, int act)
{
  __shared__ __align__(16) __bf16 sA[2][TILE_M * LDSS];
  __shared__ __align__(16) __bf16 sB[2][TILE_N * LDSS];

  const int tid   = threadIdx.x;
  const int mBase = blockIdx.y * TILE_M;
  const int nBase = blockIdx.x * TILE_N;
  const int wave  = tid >> 5;
  const int lane  = tid & 31;
  const int r     = lane & 15;        // row/col within 16x16 tile
  const int half  = lane >> 4;        // which K-half this lane carries
  const int wm    = (wave >> 1) * 64; // wave M offset (2 wave rows)
  const int wn    = (wave & 1) * 64;  // wave N offset (2 wave cols)

  v8f acc[4][4];
#pragma unroll
  for (int mi = 0; mi < 4; ++mi)
#pragma unroll
    for (int ni = 0; ni < 4; ++ni)
#pragma unroll
      for (int e = 0; e < 8; ++e) acc[mi][ni][e] = 0.0f;

  // Stage one 128x32 tile pair via async b128 chunks (8 bf16 = 16B each).
  // 512 chunks per operand, 4 per thread per operand.
  auto stage = [&](int buf, int k0) {
#pragma unroll
    for (int c = tid; c < 512; c += 128) {
      int row = c >> 2, ko = (c & 3) * 8;
      async_b128(lds_off(&sA[buf][row * LDSS + ko]),
                 A  + (size_t)(mBase + row) * K + k0 + ko);
      async_b128(lds_off(&sB[buf][row * LDSS + ko]),
                 BT + (size_t)(nBase + row) * K + k0 + ko);
    }
  };

  stage(0, 0);
  wait_async0();
  __syncthreads();

  int buf = 0;
  for (int k0 = 0; k0 < K; k0 += TILE_K) {
    if (k0 + TILE_K < K) stage(buf ^ 1, k0 + TILE_K);  // overlap with WMMA

    v16bf bfrag[4];
#pragma unroll
    for (int ni = 0; ni < 4; ++ni)
      bfrag[ni] = load_frag(&sB[buf][(wn + ni * 16 + r) * LDSS + half * 8]);
#pragma unroll
    for (int mi = 0; mi < 4; ++mi) {
      v16bf afrag = load_frag(&sA[buf][(wm + mi * 16 + r) * LDSS + half * 8]);
#pragma unroll
      for (int ni = 0; ni < 4; ++ni)
        acc[mi][ni] = __builtin_amdgcn_wmma_f32_16x16x32_bf16(
            false, afrag, false, bfrag[ni], (short)0, acc[mi][ni], false, false);
    }

    wait_async0();      // next tile resident in other buffer
    __syncthreads();    // all waves done reading current buffer
    buf ^= 1;
  }

  // Epilogue: bias + optional exact GELU. D layout: VGPR i -> row i+8*half, col r.
#pragma unroll
  for (int mi = 0; mi < 4; ++mi)
#pragma unroll
    for (int ni = 0; ni < 4; ++ni) {
      int col = nBase + wn + ni * 16 + r;
      float bv = bias ? bias[col] : 0.0f;
#pragma unroll
      for (int i = 0; i < 8; ++i) {
        int row = mBase + wm + mi * 16 + half * 8 + i;
        float v = acc[mi][ni][i] + bv;
        if (act) v = 0.5f * v * (1.0f + erff(v * 0.70710678118654752f));
        C[(size_t)row * N + col] = v;
      }
    }
}

// ---------------------------------------------------------------------------
// fp32 -> bf16 conversions
// ---------------------------------------------------------------------------
__global__ void conv_bf16(const float* __restrict__ s, __bf16* __restrict__ d, int n4) {
  int i = blockIdx.x * 256 + threadIdx.x;
  if (i < n4) {
    float4 v = ((const float4*)s)[i];
    bf16x4 o;
    o[0] = (__bf16)v.x; o[1] = (__bf16)v.y; o[2] = (__bf16)v.z; o[3] = (__bf16)v.w;
    ((bf16x4*)d)[i] = o;
  }
}

// Weight transpose+convert: src fp32 [K][N] -> dst bf16 [N][K]
__global__ __launch_bounds__(256) void wtrans_bf16(
    const float* __restrict__ src, __bf16* __restrict__ dst, int K, int N)
{
  __shared__ float t[32][33];
  int k0 = blockIdx.y * 32, n0 = blockIdx.x * 32;
  int tx = threadIdx.x & 31, ty = threadIdx.x >> 5;  // 8 rows of 32
#pragma unroll
  for (int i = 0; i < 32; i += 8)
    t[ty + i][tx] = src[(size_t)(k0 + ty + i) * N + n0 + tx];
  __syncthreads();
#pragma unroll
  for (int i = 0; i < 32; i += 8)
    dst[(size_t)(n0 + ty + i) * K + k0 + tx] = (__bf16)t[tx][ty + i];
}

// ---------------------------------------------------------------------------
// ProbSparse attention helpers (fp32, L2-resident, <1% of FLOPs)
// ---------------------------------------------------------------------------
__device__ __forceinline__ unsigned hash_u32(unsigned x) {
  x ^= x >> 16; x *= 0x7feb352dU; x ^= x >> 15; x *= 0x846ca68bU; x ^= x >> 16;
  return x;
}

__global__ void fill_idx(int* idx, int n, unsigned seed, int mod) {
  int i = blockIdx.x * 256 + threadIdx.x;
  if (i < n) idx[i] = (int)(hash_u32(seed + (unsigned)i) % (unsigned)mod);
}

__global__ __launch_bounds__(32) void sample_m(
    const float* __restrict__ Q, const float* __restrict__ Kc,
    const int* __restrict__ idx, float* __restrict__ Mout)
{
  int bid = blockIdx.x;
  int l  = bid % SEQ;
  int bh = bid / SEQ;
  int h  = bh % NH, b = bh / NH;
  int t  = threadIdx.x;
  size_t qb = (size_t)(b * SEQ + l) * DM + h * DK;
  float q0 = Q[qb + t], q1 = Q[qb + 32 + t];
  float mx = -1e30f, sum = 0.0f;
  for (int s = 0; s < SAMPLEK; ++s) {
    int kl = idx[l * SAMPLEK + s];
    size_t kb = (size_t)(b * SEQ + kl) * DM + h * DK;
    float p = q0 * Kc[kb + t] + q1 * Kc[kb + 32 + t];
#pragma unroll
    for (int m = 16; m > 0; m >>= 1) p += __shfl_xor(p, m, 32);
    mx = fmaxf(mx, p);
    sum += p;
  }
  if (t == 0) Mout[(size_t)bh * SEQ + l] = mx - sum / (float)SEQ;
}

__global__ __launch_bounds__(256) void topk_sel(
    const float* __restrict__ Mv, int* __restrict__ topidx)
{
  int bh = blockIdx.x;
  __shared__ float vals[SEQ];
  __shared__ float rmax[256];
  __shared__ int   rarg[256];
  for (int i = threadIdx.x; i < SEQ; i += 256) vals[i] = Mv[(size_t)bh * SEQ + i];
  __syncthreads();
  for (int u = 0; u < NTOP; ++u) {
    float best = -1e30f; int bi = 0;
    for (int i = threadIdx.x; i < SEQ; i += 256)
      if (vals[i] > best) { best = vals[i]; bi = i; }
    rmax[threadIdx.x] = best; rarg[threadIdx.x] = bi;
    __syncthreads();
    for (int st = 128; st > 0; st >>= 1) {
      if (threadIdx.x < st && rmax[threadIdx.x + st] > rmax[threadIdx.x]) {
        rmax[threadIdx.x] = rmax[threadIdx.x + st];
        rarg[threadIdx.x] = rarg[threadIdx.x + st];
      }
      __syncthreads();
    }
    if (threadIdx.x == 0) { topidx[bh * NTOP + u] = rarg[0]; vals[rarg[0]] = -1e30f; }
    __syncthreads();
  }
}

__global__ __launch_bounds__(64) void vmean_kernel(
    const float* __restrict__ V, float* __restrict__ vm)
{
  int bh = blockIdx.x, h = bh % NH, b = bh / NH, d = threadIdx.x;
  float s = 0.0f;
  for (int s2 = 0; s2 < SEQ; ++s2)
    s += V[(size_t)(b * SEQ + s2) * DM + h * DK + d];
  vm[bh * DK + d] = s * (1.0f / (float)SEQ);
}

__global__ void ctx_fill(const float* __restrict__ vm, float* __restrict__ ctx) {
  size_t i = (size_t)blockIdx.x * 256 + threadIdx.x;
  if (i < (size_t)ROWS * DM) {
    int f = (int)(i % DM), h = f / DK, d = f % DK;
    int b = (int)(i / DM) / SEQ;
    ctx[i] = vm[(b * NH + h) * DK + d];
  }
}

__global__ __launch_bounds__(256) void attn_update(
    const float* __restrict__ Q, const float* __restrict__ Kc,
    const float* __restrict__ V, const int* __restrict__ topidx,
    float* __restrict__ ctx)
{
  int blk = blockIdx.x;
  int u = blk % NTOP, bh = blk / NTOP, h = bh % NH, b = bh / NH;
  int l = topidx[bh * NTOP + u];

  __shared__ float sc[SEQ];
  __shared__ float red[256];
  __shared__ float qv[DK];

  if (threadIdx.x < DK)
    qv[threadIdx.x] = Q[(size_t)(b * SEQ + l) * DM + h * DK + threadIdx.x];
  __syncthreads();

  float lmax = -1e30f;
  for (int k = threadIdx.x; k < SEQ; k += 256) {
    const float* kp = &Kc[(size_t)(b * SEQ + k) * DM + h * DK];
    float dot = 0.0f;
#pragma unroll
    for (int d = 0; d < DK; ++d) dot += qv[d] * kp[d];
    dot *= 0.125f;  // 1/sqrt(64)
    sc[k] = dot;
    lmax = fmaxf(lmax, dot);
  }
  red[threadIdx.x] = lmax; __syncthreads();
  for (int st = 128; st > 0; st >>= 1) {
    if (threadIdx.x < st) red[threadIdx.x] = fmaxf(red[threadIdx.x], red[threadIdx.x + st]);
    __syncthreads();
  }
  float mx = red[0]; __syncthreads();

  float lsum = 0.0f;
  for (int k = threadIdx.x; k < SEQ; k += 256) {
    float e = __expf(sc[k] - mx);
    sc[k] = e; lsum += e;
  }
  red[threadIdx.x] = lsum; __syncthreads();
  for (int st = 128; st > 0; st >>= 1) {
    if (threadIdx.x < st) red[threadIdx.x] += red[threadIdx.x + st];
    __syncthreads();
  }
  float inv = 1.0f / red[0]; __syncthreads();

  int d = threadIdx.x & 63, g = threadIdx.x >> 6;
  float accum = 0.0f;
  for (int k = g * 256; k < g * 256 + 256; ++k)
    accum += sc[k] * V[(size_t)(b * SEQ + k) * DM + h * DK + d];
  red[threadIdx.x] = accum; __syncthreads();
  if (threadIdx.x < DK) {
    float t = red[threadIdx.x] + red[64 + threadIdx.x] +
              red[128 + threadIdx.x] + red[192 + threadIdx.x];
    ctx[(size_t)(b * SEQ + l) * DM + h * DK + threadIdx.x] = t * inv;
  }
}

// ---------------------------------------------------------------------------
// Elementwise + BatchNorm
// ---------------------------------------------------------------------------
__global__ void add_kernel(const float* __restrict__ a, const float* __restrict__ b,
                           float* __restrict__ o, int n) {
  int i = blockIdx.x * 256 + threadIdx.x;
  if (i < n) o[i] = a[i] + b[i];
}

__global__ __launch_bounds__(256) void bn_stats(
    const float* __restrict__ s, float* __restrict__ mu, float* __restrict__ rs,
    int rows, int cols)
{
  int f = blockIdx.x;
  float sum = 0.0f, sq = 0.0f;
  for (int r = threadIdx.x; r < rows; r += 256) {
    float v = s[(size_t)r * cols + f];
    sum += v; sq += v * v;
  }
  __shared__ float ls[256], lq[256];
  ls[threadIdx.x] = sum; lq[threadIdx.x] = sq; __syncthreads();
  for (int st = 128; st > 0; st >>= 1) {
    if (threadIdx.x < st) { ls[threadIdx.x] += ls[threadIdx.x + st];
                            lq[threadIdx.x] += lq[threadIdx.x + st]; }
    __syncthreads();
  }
  if (threadIdx.x == 0) {
    float m = ls[0] / (float)rows;
    float var = lq[0] / (float)rows - m * m;
    mu[f] = m; rs[f] = rsqrtf(var + 1e-5f);
  }
}

__global__ void bn_apply(float* __restrict__ s, const float* __restrict__ mu,
                         const float* __restrict__ rs, const float* __restrict__ g,
                         const float* __restrict__ bb, int n, int cols) {
  int i = blockIdx.x * 256 + threadIdx.x;
  if (i < n) {
    int f = i % cols;
    s[i] = (s[i] - mu[f]) * rs[f] * g[f] + bb[f];
  }
}

// ---------------------------------------------------------------------------
// Orchestration
// ---------------------------------------------------------------------------
extern "C" void kernel_launch(void* const* d_in, const int* in_sizes, int n_in,
                              void* d_out, int out_size, void* d_ws, size_t ws_size,
                              hipStream_t stream) {
  (void)in_sizes; (void)n_in; (void)out_size; (void)ws_size;

  const float* tgt  = (const float*)d_in[0];
  const float* mem  = (const float*)d_in[1];
  const float* sa_wq = (const float*)d_in[2];  const float* sa_bq = (const float*)d_in[3];
  const float* sa_wk = (const float*)d_in[4];  const float* sa_bk = (const float*)d_in[5];
  const float* sa_wv = (const float*)d_in[6];  const float* sa_bv = (const float*)d_in[7];
  const float* sa_wo = (const float*)d_in[8];  const float* sa_bo = (const float*)d_in[9];
  const float* ca_wq = (const float*)d_in[10]; const float* ca_bq = (const float*)d_in[11];
  const float* ca_wk = (const float*)d_in[12]; const float* ca_bk = (const float*)d_in[13];
  const float* ca_wv = (const float*)d_in[14]; const float* ca_bv = (const float*)d_in[15];
  const float* ca_wo = (const float*)d_in[16]; const float* ca_bo = (const float*)d_in[17];
  const float* w1 = (const float*)d_in[18]; const float* b1 = (const float*)d_in[19];
  const float* w2 = (const float*)d_in[20]; const float* b2 = (const float*)d_in[21];
  const float* bn1_g = (const float*)d_in[22]; const float* bn1_b = (const float*)d_in[23];
  const float* bn2_g = (const float*)d_in[24]; const float* bn2_b = (const float*)d_in[25];
  const float* bn3_g = (const float*)d_in[26]; const float* bn3_b = (const float*)d_in[27];

  // Workspace carve
  char* w = (char*)d_ws;
  auto carve = [&](size_t bytes) -> void* {
    void* p = (void*)w;
    w += (bytes + 255) & ~(size_t)255;
    return p;
  };
  const size_t MAT  = (size_t)ROWS * DM * sizeof(float);        // 16 MB fp32
  const size_t MATH = (size_t)ROWS * DM * sizeof(__bf16);       //  8 MB bf16
  float*  Qb  = (float*)carve(MAT);
  float*  Kb  = (float*)carve(MAT);
  float*  Vb  = (float*)carve(MAT);
  float*  CTX = (float*)carve(MAT);
  float*  T2  = (float*)carve(MAT);
  float*  X   = (float*)carve(MAT);
  float*  X2  = (float*)carve(MAT);
  float*  Hf  = (float*)carve((size_t)ROWS * DFF_N * sizeof(float));  // 64 MB
  __bf16* AbBig = (__bf16*)carve((size_t)ROWS * DFF_N * sizeof(__bf16)); // 32 MB
  __bf16* Ab2   = (__bf16*)carve(MATH);                         // 8 MB
  __bf16* Wq  = (__bf16*)carve((size_t)DM * DM * sizeof(__bf16));
  __bf16* Wk  = (__bf16*)carve((size_t)DM * DM * sizeof(__bf16));
  __bf16* Wv  = (__bf16*)carve((size_t)DM * DM * sizeof(__bf16));
  __bf16* Wo  = (__bf16*)carve((size_t)DM * DM * sizeof(__bf16));
  __bf16* W1T = (__bf16*)carve((size_t)DM * DFF_N * sizeof(__bf16));
  __bf16* W2T = (__bf16*)carve((size_t)DM * DFF_N * sizeof(__bf16));
  int*    idx = (int*)carve((size_t)SEQ * SAMPLEK * sizeof(int));
  float*  Mb  = (float*)carve((size_t)BATCH * NH * SEQ * sizeof(float));
  int*    ti  = (int*)carve((size_t)BATCH * NH * NTOP * sizeof(int));
  float*  vm  = (float*)carve((size_t)BATCH * NH * DK * sizeof(float));
  float*  mu  = (float*)carve((size_t)DM * sizeof(float));
  float*  rs  = (float*)carve((size_t)DM * sizeof(float));

  float* OUT = (float*)d_out;
  const int ELT_BLOCKS = (ROWS * DM) / 256;  // 16384

  auto gemm = [&](const __bf16* A, const __bf16* BT, const float* bias, float* C,
                  int M, int N, int K, int act) {
    dim3 grid(N / TILE_N, M / TILE_M);
    gemm_bf16_wmma<<<grid, 128, 0, stream>>>(A, BT, bias, C, M, N, K, act);
  };
  auto conv = [&](const float* s, __bf16* d, int n) {
    conv_bf16<<<(n / 4 + 255) / 256, 256, 0, stream>>>(s, d, n / 4);
  };
  auto wtrans = [&](const float* s, __bf16* d, int K, int N) {
    dim3 grid(N / 32, K / 32);
    wtrans_bf16<<<grid, 256, 0, stream>>>(s, d, K, N);
  };

  auto prob_attn = [&](const __bf16* qA /*bf16 query input*/, const __bf16* kvA,
                       const float* bq, const float* bk, const float* bv,
                       const float* bo, unsigned seed, float* out_t2) {
    gemm(qA,  Wq, bq, Qb, ROWS, DM, DM, 0);
    gemm(kvA, Wk, bk, Kb, ROWS, DM, DM, 0);
    gemm(kvA, Wv, bv, Vb, ROWS, DM, DM, 0);
    fill_idx<<<(SEQ * SAMPLEK + 255) / 256, 256, 0, stream>>>(idx, SEQ * SAMPLEK, seed, SEQ);
    sample_m<<<BATCH * NH * SEQ, 32, 0, stream>>>(Qb, Kb, idx, Mb);
    topk_sel<<<BATCH * NH, 256, 0, stream>>>(Mb, ti);
    vmean_kernel<<<BATCH * NH, 64, 0, stream>>>(Vb, vm);
    ctx_fill<<<ELT_BLOCKS, 256, 0, stream>>>(vm, CTX);
    attn_update<<<BATCH * NH * NTOP, 256, 0, stream>>>(Qb, Kb, Vb, ti, CTX);
    conv(CTX, Ab2, ROWS * DM);
    gemm(Ab2, Wo, bo, out_t2, ROWS, DM, DM, 0);
  };

  // ---- Self-attention block ----
  wtrans(sa_wq, Wq, DM, DM); wtrans(sa_wk, Wk, DM, DM);
  wtrans(sa_wv, Wv, DM, DM); wtrans(sa_wo, Wo, DM, DM);
  conv(tgt, AbBig, ROWS * DM);
  prob_attn(AbBig, AbBig, sa_bq, sa_bk, sa_bv, sa_bo, 0x13579BDFu, T2);
  add_kernel<<<ELT_BLOCKS, 256, 0, stream>>>(tgt, T2, X, ROWS * DM);
  bn_stats<<<DM, 256, 0, stream>>>(X, mu, rs, ROWS, DM);
  bn_apply<<<ELT_BLOCKS, 256, 0, stream>>>(X, mu, rs, bn1_g, bn1_b, ROWS * DM, DM);

  // ---- Cross-attention block ----
  wtrans(ca_wq, Wq, DM, DM); wtrans(ca_wk, Wk, DM, DM);
  wtrans(ca_wv, Wv, DM, DM); wtrans(ca_wo, Wo, DM, DM);
  conv(X, AbBig, ROWS * DM);          // query input
  __bf16* MemB = AbBig + (size_t)ROWS * DM;   // second 8MB slice of AbBig
  conv(mem, MemB, ROWS * DM);         // key/value input
  prob_attn(AbBig, MemB, ca_bq, ca_bk, ca_bv, ca_bo, 0x9E3779B9u, T2);
  add_kernel<<<ELT_BLOCKS, 256, 0, stream>>>(X, T2, X2, ROWS * DM);
  bn_stats<<<DM, 256, 0, stream>>>(X2, mu, rs, ROWS, DM);
  bn_apply<<<ELT_BLOCKS, 256, 0, stream>>>(X2, mu, rs, bn2_g, bn2_b, ROWS * DM, DM);

  // ---- FFN block ----
  wtrans(w1, W1T, DM, DFF_N);
  wtrans(w2, W2T, DFF_N, DM);
  conv(X2, AbBig, ROWS * DM);
  gemm(AbBig, W1T, b1, Hf, ROWS, DFF_N, DM, 1);      // fused exact GELU
  conv(Hf, AbBig, ROWS * DFF_N);                      // 32 MB bf16
  gemm(AbBig, W2T, b2, T2, ROWS, DM, DFF_N, 0);
  add_kernel<<<ELT_BLOCKS, 256, 0, stream>>>(X2, T2, OUT, ROWS * DM);
  bn_stats<<<DM, 256, 0, stream>>>(OUT, mu, rs, ROWS, DM);
  bn_apply<<<ELT_BLOCKS, 256, 0, stream>>>(OUT, mu, rs, bn3_g, bn3_b, ROWS * DM, DM);
}
